// RNNBezierAE_55070070669744
// MI455X (gfx1250) — matrix operand order repaired and checked by reading, not compiled, not run on hardware
//
#include <hip/hip_runtime.h>
#include <math.h>

#define T_   256
#define B_   512
#define NI_  2
#define H_   256
#define DEG_ 9
#define NP_  522
#define NHC_ 1024
#define HSTR 264   // padded LDS row stride (bf16 elems) to spread banks

typedef __attribute__((ext_vector_type(16))) __bf16         v16bf;
typedef __attribute__((ext_vector_type(8)))  float          v8f;
typedef __attribute__((ext_vector_type(16))) unsigned short us16;
typedef __attribute__((ext_vector_type(8)))  unsigned short us8;

union U16 { us16 u; us8 h[2]; v16bf b; };

__device__ __forceinline__ unsigned short f2bf(float f) {
  unsigned int u = __float_as_uint(f);
  unsigned int r = u + 0x7FFFu + ((u >> 16) & 1u);   // round-to-nearest-even
  return (unsigned short)(r >> 16);
}
__device__ __forceinline__ float sigf(float x) { return 1.0f / (1.0f + __expf(-x)); }

// ---------------------------------------------------------------------------
// init: t_logits <- b_t everywhere (softmax sees b_t at padded positions),
// zero the regu accumulator in d_out.
__global__ void k_init(float* tl, const float* __restrict__ bt, float* regu) {
  int i = blockIdx.x * blockDim.x + threadIdx.x;
  if (i < B_ * T_) tl[i] = bt[0];
  if (i == 0) *regu = 0.0f;
}

// ---------------------------------------------------------------------------
// pack W_hh (both dirs) into bf16 WMMA B-fragment layout:
// e = ((dir*8 + kt)*64 + nt)*512 + lane*16 + i
//   n = nt*16 + (lane&15), k = kt*32 + (lane>>4)*16 + i
__global__ void k_pack(const float* __restrict__ whf, const float* __restrict__ whb,
                       unsigned short* __restrict__ pw) {
  int e = blockIdx.x * blockDim.x + threadIdx.x;
  if (e >= 2 * 8 * 64 * 512) return;
  int i    = e & 15;
  int lane = (e >> 4) & 31;
  int nt   = (e >> 9) & 63;
  int kt   = (e >> 15) & 7;
  int dir  = e >> 18;
  int n = nt * 16 + (lane & 15);
  int k = kt * 32 + (lane >> 4) * 16 + i;
  const float* w = dir ? whb : whf;
  pw[e] = f2bf(w[(size_t)n * H_ + k]);
}

// ---------------------------------------------------------------------------
// Recurrent bidirectional LSTM. One workgroup = 16 batch rows, 8 waves.
// Wave w owns H-columns [w*32, w*32+32): 8 WMMA col-tiles (i,f,g,o × 2).
// h kept in LDS as bf16; c and frozen-h kept in registers (f32).
// W_hh fragments are streamed from L2 every step. A zero integer offset,
// laundered through a volatile asm each step, blocks LICM from hoisting the
// t-invariant fragment loads (which would spill), while keeping the pointer's
// global address-space provenance (global_load, LOADcnt only).
__global__ __launch_bounds__(256) void k_lstm(
    const float* __restrict__ x, const float* __restrict__ h0g,
    const float* __restrict__ c0g, const int* __restrict__ lengths,
    const float* __restrict__ wihf, const float* __restrict__ bihf, const float* __restrict__ bhhf,
    const float* __restrict__ wihb, const float* __restrict__ bihb, const float* __restrict__ bhhb,
    const float* __restrict__ wt, const unsigned short* __restrict__ packW,
    float* __restrict__ tl, float* __restrict__ hT, float* __restrict__ cT) {
  __shared__ __align__(16) unsigned short hlds[16 * HSTR];
  __shared__ float xs[16 * 2];
  __shared__ float tlpart[16];
  __shared__ int   lens[16];

  const int tid  = threadIdx.x;
  const int dir  = blockIdx.z;
  const int b0   = blockIdx.x * 16;
  const int wid  = tid >> 5;
  const int lane = tid & 31;
  const int hlf  = lane >> 4;       // lane half (C/D rows 0-7 vs 8-15)
  const int colw = lane & 15;       // N within a 16-col tile
  const int j0   = wid * 32;        // wave's H-column base
  const int arow = lane & 15;       // A-fragment row
  const int aklo = hlf * 16;        // A-fragment K sub-offset

  const float* wih = dir ? wihb : wihf;
  const float* bih = dir ? bihb : bihf;
  const float* bhh = dir ? bhhb : bhhf;

  // stage initial h (bf16) and lengths
  for (int idx = tid; idx < 16 * H_; idx += 256) {
    int row = idx >> 8, col = idx & 255;
    hlds[row * HSTR + col] = f2bf(h0g[((size_t)(dir * B_ + b0 + row)) * H_ + col]);
  }
  if (tid < 16) lens[tid] = lengths[b0 + tid];
  __syncthreads();

  // per-lane constants: x-projection weights + fused bias per gate tile, w_t slice
  float wx0[4][2], wx1[4][2], bsum[4][2], wtv[2];
  float cst[2][8], hst[2][8];
  int lenr[8];
#pragma unroll
  for (int g = 0; g < 4; ++g)
#pragma unroll
    for (int s2 = 0; s2 < 2; ++s2) {
      int gc = g * H_ + j0 + s2 * 16 + colw;
      wx0[g][s2]  = wih[gc * NI_ + 0];
      wx1[g][s2]  = wih[gc * NI_ + 1];
      bsum[g][s2] = bih[gc] + bhh[gc];
    }
#pragma unroll
  for (int s2 = 0; s2 < 2; ++s2) wtv[s2] = wt[dir * H_ + j0 + s2 * 16 + colw];
#pragma unroll
  for (int s2 = 0; s2 < 2; ++s2)
#pragma unroll
    for (int r = 0; r < 8; ++r) {
      int row = r + hlf * 8;
      int col = j0 + s2 * 16 + colw;
      cst[s2][r] = c0g[((size_t)(dir * B_ + b0 + row)) * H_ + col];
      hst[s2][r] = h0g[((size_t)(dir * B_ + b0 + row)) * H_ + col];
    }
#pragma unroll
  for (int r = 0; r < 8; ++r) lenr[r] = lens[r + hlf * 8];

  const unsigned short* wpd = packW + (size_t)dir * 8 * 64 * 512;

  for (int t = 0; t < T_; ++t) {
    if (tid < 16) {
      int L  = lens[tid];
      int tt = dir ? ((L - 1 - t) > 0 ? (L - 1 - t) : 0) : t;
      xs[tid * 2 + 0] = x[((size_t)tt * B_ + b0 + tid) * NI_ + 0];
      xs[tid * 2 + 1] = x[((size_t)tt * B_ + b0 + tid) * NI_ + 1];
      tlpart[tid] = 0.0f;
    }
    __syncthreads();   // (a) xs + h_{t-1} stable

    // Opaque zero offset, redefined every step: blocks LICM across the time
    // loop but preserves the global address space of the weight pointer.
    size_t topaque = 0;
    asm volatile("" : "+s"(topaque));
    const unsigned short* wbase = wpd + topaque;

    v8f acc[8];
#pragma unroll
    for (int u = 0; u < 8; ++u) acc[u] = (v8f){0.f, 0.f, 0.f, 0.f, 0.f, 0.f, 0.f, 0.f};

    for (int kt = 0; kt < 8; ++kt) {
      // A fragment (h tile) from LDS: two b128 ds loads
      U16 au;
      const unsigned short* hp = &hlds[arow * HSTR + kt * 32 + aklo];
      au.h[0] = *(const us8*)(hp);
      au.h[1] = *(const us8*)(hp + 8);

      const unsigned short* wp = wbase + (size_t)kt * 64 * 512 + lane * 16;
      // two half-iterations: load 4 B-fragments as a clause, then 4 WMMAs
#pragma unroll
      for (int g2 = 0; g2 < 2; ++g2) {
        U16 bu[4];
#pragma unroll
        for (int q = 0; q < 4; ++q) {
          int tile = g2 * 4 + q;
          int g = tile >> 1, s2 = tile & 1;
          int nt = g * 16 + wid * 2 + s2;
          bu[q].u = *(const us16*)(wp + (size_t)nt * 512);
        }
#pragma unroll
        for (int q = 0; q < 4; ++q) {
          int tile = g2 * 4 + q;
          acc[tile] = __builtin_amdgcn_wmma_f32_16x16x32_bf16(
              false, au.b, false, bu[q].b, (short)0, acc[tile], false, false);
        }
      }
    }
    __syncthreads();   // (b) all hlds reads done before rewrite

    // gates -> state update -> new h into LDS, partial dot with w_t
#pragma unroll
    for (int r = 0; r < 8; ++r) {
      int row   = r + hlf * 8;
      float x0v = xs[row * 2 + 0], x1v = xs[row * 2 + 1];
      bool  m   = (t < lenr[r]);
      float pr  = 0.0f;
#pragma unroll
      for (int s2 = 0; s2 < 2; ++s2) {
        float iv = sigf(acc[0 + s2][r] + x0v * wx0[0][s2] + x1v * wx1[0][s2] + bsum[0][s2]);
        float fv = sigf(acc[2 + s2][r] + x0v * wx0[1][s2] + x1v * wx1[1][s2] + bsum[1][s2]);
        float gv = tanhf(acc[4 + s2][r] + x0v * wx0[2][s2] + x1v * wx1[2][s2] + bsum[2][s2]);
        float ov = sigf(acc[6 + s2][r] + x0v * wx0[3][s2] + x1v * wx1[3][s2] + bsum[3][s2]);
        float cn = fv * cst[s2][r] + iv * gv;
        float hn = ov * tanhf(cn);
        cst[s2][r] = m ? cn : cst[s2][r];
        hst[s2][r] = m ? hn : hst[s2][r];
        hlds[row * HSTR + j0 + s2 * 16 + colw] = f2bf(hst[s2][r]);
        pr += (m ? hn : 0.0f) * wtv[s2];
      }
      pr += __shfl_xor(pr, 1, 16);
      pr += __shfl_xor(pr, 2, 16);
      pr += __shfl_xor(pr, 4, 16);
      pr += __shfl_xor(pr, 8, 16);
      if (colw == 0) atomicAdd(&tlpart[row], pr);
    }
    __syncthreads();   // (c) new h + tlpart complete

    if (tid < 16) {
      int L   = lens[tid];
      int pos = dir ? ((L - 1 - t) > 0 ? (L - 1 - t) : 0) : t;
      atomicAdd(&tl[(size_t)(b0 + tid) * T_ + pos], tlpart[tid]);
    }
  }

  // final (frozen) states
#pragma unroll
  for (int s2 = 0; s2 < 2; ++s2)
#pragma unroll
    for (int r = 0; r < 8; ++r) {
      int row = r + hlf * 8;
      int col = j0 + s2 * 16 + colw;
      hT[((size_t)(dir * B_ + b0 + row)) * H_ + col] = hst[s2][r];
      cT[((size_t)(dir * B_ + b0 + row)) * H_ + col] = cst[s2][r];
    }
}

// ---------------------------------------------------------------------------
// softmax over T then inclusive cumsum -> ts[b,t]; one block per batch row
__global__ __launch_bounds__(256) void k_softcum(const float* __restrict__ tl,
                                                 float* __restrict__ ts) {
  __shared__ float sh[256];
  int b = blockIdx.x, t = threadIdx.x;
  float v = tl[(size_t)b * T_ + t];
  sh[t] = v;
  __syncthreads();
  for (int s = 128; s > 0; s >>= 1) {
    if (t < s) sh[t] = fmaxf(sh[t], sh[t + s]);
    __syncthreads();
  }
  float mx = sh[0];
  __syncthreads();
  float e = __expf(v - mx);
  sh[t] = e;
  __syncthreads();
  for (int off = 1; off < 256; off <<= 1) {
    float add = (t >= off) ? sh[t - off] : 0.0f;
    __syncthreads();
    sh[t] += add;
    __syncthreads();
  }
  float total = sh[255];
  ts[(size_t)b * T_ + t] = sh[t] / total;
}

// ---------------------------------------------------------------------------
// heads: proj = relu(W_proj @ [hTf,hTb,cTf,cTf] + b), control points P,
// rational weights R = sigmoid([0,rw,0]), regu accumulation.
__global__ __launch_bounds__(256) void k_head(
    const float* __restrict__ hT, const float* __restrict__ cT,
    const float* __restrict__ wproj, const float* __restrict__ bproj,
    const float* __restrict__ wctrl, const float* __restrict__ bctrl,
    const float* __restrict__ wratw, const float* __restrict__ bratw,
    float* __restrict__ Pws, float* __restrict__ Rws, float* __restrict__ regu) {
  __shared__ float hc[NHC_];
  __shared__ float pj[NP_];
  __shared__ float pl[20];
  int b = blockIdx.x, tid = threadIdx.x;
  for (int i = tid; i < H_; i += 256) {
    hc[i]          = hT[((size_t)(0 * B_ + b)) * H_ + i];
    hc[H_ + i]     = hT[((size_t)(1 * B_ + b)) * H_ + i];
    float cf       = cT[((size_t)(0 * B_ + b)) * H_ + i];
    hc[2 * H_ + i] = cf;
    hc[3 * H_ + i] = cf;   // cTf used twice (faithful to reference)
  }
  __syncthreads();
  for (int n = tid; n < NP_; n += 256) {
    float a = bproj[n];
    const float* wr = wproj + (size_t)n * NHC_;
    for (int k = 0; k < NHC_; ++k) a += wr[k] * hc[k];
    pj[n] = fmaxf(a, 0.0f);
  }
  __syncthreads();
  if (tid < 20) {
    float a = bctrl[tid];
    const float* wr = wctrl + (size_t)tid * NP_;
    for (int k = 0; k < NP_; ++k) a += wr[k] * pj[k];
    Pws[(size_t)b * 20 + tid] = a;
    pl[tid] = a;
  } else if (tid < 28) {
    int n = tid - 20;
    float a = bratw[n];
    const float* wr = wratw + (size_t)n * NP_;
    for (int k = 0; k < NP_; ++k) a += wr[k] * pj[k];
    Rws[(size_t)b * 10 + 1 + n] = sigf(a);
  } else if (tid == 28) {
    Rws[(size_t)b * 10 + 0] = 0.5f;
    Rws[(size_t)b * 10 + 9] = 0.5f;
  }
  __syncthreads();
  if (tid == 0) {
    float s = 0.0f;
    for (int i = 0; i < DEG_; ++i) {
      float dx = pl[2 * (i + 1)] - pl[2 * i];
      float dy = pl[2 * (i + 1) + 1] - pl[2 * i + 1];
      s += dx * dx + dy * dy;
    }
    atomicAdd(regu, s * (1.0f / (9.0f * (float)B_)));
  }
}

// ---------------------------------------------------------------------------
// rational Bézier evaluation; one block per batch row, one thread per t
__global__ __launch_bounds__(256) void k_bezier(
    const float* __restrict__ ts, const float* __restrict__ Pws,
    const float* __restrict__ Rws, const int* __restrict__ lengths,
    float* __restrict__ out) {
  __shared__ float Pl[20];
  __shared__ float Rl[10];
  int b = blockIdx.x, t = threadIdx.x;
  if (t < 20) Pl[t] = Pws[(size_t)b * 20 + t];
  if (t < 10) Rl[t] = Rws[(size_t)b * 10 + t];
  __syncthreads();
  float tv = ts[(size_t)b * T_ + t];
  float sv = 1.0f - tv;
  float tp[DEG_ + 1], sp[DEG_ + 1];
  tp[0] = 1.0f; sp[0] = 1.0f;
#pragma unroll
  for (int i = 1; i <= DEG_; ++i) { tp[i] = tp[i - 1] * tv; sp[i] = sp[i - 1] * sv; }
  const float binom[10] = {1.f, 9.f, 36.f, 84.f, 126.f, 126.f, 84.f, 36.f, 9.f, 1.f};
  float wsum = 0.f, n0 = 0.f, n1 = 0.f;
#pragma unroll
  for (int i = 0; i <= DEG_; ++i) {
    float w = binom[i] * tp[i] * sp[DEG_ - i] * Rl[i];
    wsum += w;
    n0 += w * Pl[2 * i];
    n1 += w * Pl[2 * i + 1];
  }
  float m   = (t < lengths[b]) ? 1.0f : 0.0f;
  float inv = m / wsum;
  out[((size_t)b * T_ + t) * 2 + 0] = n0 * inv;
  out[((size_t)b * T_ + t) * 2 + 1] = n1 * inv;
}

// ---------------------------------------------------------------------------
extern "C" void kernel_launch(void* const* d_in, const int* in_sizes, int n_in,
                              void* d_out, int out_size, void* d_ws, size_t ws_size,
                              hipStream_t stream) {
  const float* x     = (const float*)d_in[0];
  const float* h0    = (const float*)d_in[1];
  const float* c0    = (const float*)d_in[2];
  const int*   len   = (const int*)d_in[3];
  const float* wihf  = (const float*)d_in[4];
  const float* whhf  = (const float*)d_in[5];
  const float* bihf  = (const float*)d_in[6];
  const float* bhhf  = (const float*)d_in[7];
  const float* wihb  = (const float*)d_in[8];
  const float* whhb  = (const float*)d_in[9];
  const float* bihb  = (const float*)d_in[10];
  const float* bhhb  = (const float*)d_in[11];
  const float* wt    = (const float*)d_in[12];
  const float* bt    = (const float*)d_in[13];
  const float* wproj = (const float*)d_in[14];
  const float* bproj = (const float*)d_in[15];
  const float* wctrl = (const float*)d_in[16];
  const float* bctrl = (const float*)d_in[17];
  const float* wratw = (const float*)d_in[18];
  const float* bratw = (const float*)d_in[19];
  float* out = (float*)d_out;

  char* ws = (char*)d_ws;
  unsigned short* packW = (unsigned short*)(ws);                       // 1 MB
  float* tl  = (float*)(ws + (1u << 20));                              // 512 KB
  float* tsb = (float*)(ws + (1u << 20) + (512u << 10));               // 512 KB
  float* hT  = (float*)(ws + (2u << 20));                              // 1 MB
  float* cT  = (float*)(ws + (3u << 20));                              // 1 MB
  float* Pws = (float*)(ws + (4u << 20));                              // 40 KB
  float* Rws = Pws + (size_t)B_ * 20;                                  // 20 KB
  float* regu = out + (size_t)B_ * T_ * 2;

  k_init<<<(B_ * T_ + 255) / 256, 256, 0, stream>>>(tl, bt, regu);
  k_pack<<<(2 * 8 * 64 * 512 + 255) / 256, 256, 0, stream>>>(whhf, whhb, packW);
  dim3 lg(B_ / 16, 1, 2);
  k_lstm<<<lg, 256, 0, stream>>>(x, h0, c0, len, wihf, bihf, bhhf,
                                 wihb, bihb, bhhb, wt, packW, tl, hT, cT);
  k_softcum<<<B_, 256, 0, stream>>>(tl, tsb);
  k_head<<<B_, 256, 0, stream>>>(hT, cT, wproj, bproj, wctrl, bctrl,
                                 wratw, bratw, Pws, Rws, regu);
  k_bezier<<<B_, 256, 0, stream>>>(tsb, Pws, Rws, len, out);
}
